// DONEBase_43310450213578
// MI455X (gfx1250) — matrix operand order repaired and checked by reading, not compiled, not run on hardware
//
#include <hip/hip_runtime.h>
#include <hip/hip_bf16.h>

typedef __attribute__((ext_vector_type(16))) __bf16 v16bf;
typedef __attribute__((ext_vector_type(2)))  __bf16 v2bf;
typedef __attribute__((ext_vector_type(8)))  float  v8f;

#define BM 64
#define BN 64
#define BK 32
#define LDS_STRIDE 36   // 32 + 4 ushorts pad: 4B-aligned rows, spread banks

// Let the backend pick the native convert (v_cvt_pk_bf16_f32 class) if present.
__device__ __forceinline__ unsigned short f2bf(float f) {
    __bf16 b = (__bf16)f;
    return __builtin_bit_cast(unsigned short, b);
}
__device__ __forceinline__ unsigned int f2bf2(float lo, float hi) {
    v2bf v;
    v[0] = (__bf16)lo;
    v[1] = (__bf16)hi;
    return __builtin_bit_cast(unsigned int, v);
}

// One-time: W[K,N] fp32 -> WT[N,K] bf16 (transposed so B fragments are
// contiguous 32B runs per lane, loadable with global_load_b128 from L2).
__global__ __launch_bounds__(256)
void convert_transpose_w(const float* __restrict__ W, unsigned short* __restrict__ WT,
                         int K, int N, int total)
{
    const int idx = blockIdx.x * 256 + threadIdx.x;
    if (idx < total) {
        const int k = idx / N;
        const int n = idx - k * N;
        WT[(size_t)n * K + k] = f2bf(W[idx]);
    }
}

// C[M,N] = act(A[M,K] @ W[K,N] + b[N]); A fp32 row-major, W given as WT[N,K] bf16.
// Requires M%64==0, N%64==0, K%32==0.
__global__ __launch_bounds__(256)
void gemm_bias_act(const float* __restrict__ A, const unsigned short* __restrict__ WT,
                   const float* __restrict__ bias, float* __restrict__ C,
                   int M, int N, int K, int relu)
{
    __shared__ unsigned short As[2][BM][LDS_STRIDE]; // double-buffered A tile (bf16 bits)

    const int tid  = threadIdx.x;
    const int lane = tid & 31;
    const int wid  = tid >> 5;
    const int wm   = wid >> 1;      // 0..3 : 16-row strip
    const int wn   = wid & 1;       // 0..1 : 32-col strip
    const int r    = lane & 15;
    const int half = lane >> 4;

    const int bn0 = blockIdx.x * BN;
    const int bm0 = blockIdx.y * BM;

    // A staging coords: 64x32 tile, thread -> (row, 8 cols)
    const int arow = tid >> 2;            // 0..63
    const int acol = (tid & 3) * 8;       // 0,8,16,24
    const float* arowp = A + (size_t)(bm0 + arow) * K + acol;

    // B fragment base: lane r -> column (bn0 + wn*32 + r), k run starts at 16*half
    const unsigned short* wp0 = WT + (size_t)(bn0 + wn * 32 + r) * K + 16 * half;
    const unsigned short* wp1 = wp0 + (size_t)16 * K;

    v8f acc0 = {};
    v8f acc1 = {};

    auto stageA = [&](int kbase, int buf) {
        float4 a0 = *(const float4*)(arowp + kbase);
        float4 a1 = *(const float4*)(arowp + kbase + 4);
        unsigned int* dp = (unsigned int*)&As[buf][arow][acol];
        dp[0] = f2bf2(a0.x, a0.y);
        dp[1] = f2bf2(a0.z, a0.w);
        dp[2] = f2bf2(a1.x, a1.y);
        dp[3] = f2bf2(a1.z, a1.w);
    };

    stageA(0, 0);
    __syncthreads();

    const int nTiles = K / BK;
    for (int t = 0; t < nTiles; ++t) {
        const int k0  = t * BK;
        const int buf = t & 1;

        // ---- B fragments: two 16B loads each, straight from global (L2-resident) ----
        union BF { v16bf v; uint4 q[2]; } bf0, bf1;
        bf0.q[0] = *(const uint4*)(wp0 + k0);
        bf0.q[1] = *(const uint4*)(wp0 + k0 + 8);
        bf1.q[0] = *(const uint4*)(wp1 + k0);
        bf1.q[1] = *(const uint4*)(wp1 + k0 + 8);

        // ---- stage next A tile into the other buffer while we compute ----
        if (t + 1 < nTiles) {
            stageA(k0 + BK, buf ^ 1);
            if (t + 2 < nTiles)
                __builtin_prefetch(arowp + k0 + 2 * BK, 0, 3);
        }

        // ---- A fragment from LDS: element s -> k = 16*half + s, same order as B ----
        union AF { v16bf v; unsigned int u[8]; } af;
        const int arow0 = wm * 16 + r;
        #pragma unroll
        for (int i = 0; i < 8; ++i) {
            const int ka = 2 * i + 16 * half;
            af.u[i] = *(const unsigned int*)&As[buf][arow0][ka];
        }

        acc0 = __builtin_amdgcn_wmma_f32_16x16x32_bf16(false, af.v, false, bf0.v,
                                                       (short)0, acc0, false, false);
        acc1 = __builtin_amdgcn_wmma_f32_16x16x32_bf16(false, af.v, false, bf1.v,
                                                       (short)0, acc1, false, false);
        __syncthreads();
    }

    // ---- epilogue: bias + activation, fp32 store ----
    const int col0 = bn0 + wn * 32 + r;
    const int col1 = col0 + 16;
    const float bb0 = bias[col0];
    const float bb1 = bias[col1];
    #pragma unroll
    for (int j = 0; j < 8; ++j) {
        const int row = bm0 + wm * 16 + j + 8 * half;  // C/D layout: VGPR j -> M=j (+8 hi lanes)
        float v0 = acc0[j] + bb0;
        float v1 = acc1[j] + bb1;
        if (relu) { v0 = fmaxf(v0, 0.0f); v1 = fmaxf(v1, 0.0f); }
        C[(size_t)row * N + col0] = v0;
        C[(size_t)row * N + col1] = v1;
    }
}

// One wave per edge: msg = ||h[dst]-h[src]||^2 over 128 dims, atomically
// accumulated (sum + count) at dst.
__global__ __launch_bounds__(256)
void edge_sqdist(const float* __restrict__ h, const int* __restrict__ src,
                 const int* __restrict__ dst, float* __restrict__ tot,
                 float* __restrict__ cnt, int E)
{
    const int wid  = (int)((blockIdx.x * blockDim.x + threadIdx.x) >> 5);
    const int lane = threadIdx.x & 31;
    if (wid >= E) return;
    const int si = src[wid];
    const int di = dst[wid];
    const float4 hs = ((const float4*)(h + (size_t)si * 128))[lane];
    const float4 hd = ((const float4*)(h + (size_t)di * 128))[lane];
    const float dx = hd.x - hs.x, dy = hd.y - hs.y;
    const float dz = hd.z - hs.z, dw = hd.w - hs.w;
    float acc = dx * dx + dy * dy + dz * dz + dw * dw;
    #pragma unroll
    for (int off = 16; off > 0; off >>= 1)
        acc += __shfl_xor(acc, off, 32);
    if (lane == 0) {
        atomicAdd(&tot[di], acc);
        atomicAdd(&cnt[di], 1.0f);
    }
}

__global__ __launch_bounds__(256)
void finalize_mean(const float* __restrict__ tot, const float* __restrict__ cnt,
                   float* __restrict__ out, int n)
{
    const int i = blockIdx.x * blockDim.x + threadIdx.x;
    if (i < n) {
        const float c = cnt[i];
        out[i] = (c > 0.0f) ? (tot[i] / fmaxf(c, 1.0f)) : 0.0f;
    }
}

extern "C" void kernel_launch(void* const* d_in, const int* in_sizes, int n_in,
                              void* d_out, int out_size, void* d_ws, size_t ws_size,
                              hipStream_t stream) {
    const int N_   = 8192;
    const int XD   = 256;
    const int HID  = 128;
    const int SD   = 8192;
    const int E    = in_sizes[2] / 2;

    const float* x     = (const float*)d_in[0];
    const float* s     = (const float*)d_in[1];
    const int*   ei    = (const int*)d_in[2];
    const int*   src   = ei;
    const int*   dst   = ei + E;
    const float* ae_W1 = (const float*)d_in[3];  const float* ae_b1 = (const float*)d_in[4];
    const float* ae_W2 = (const float*)d_in[5];  const float* ae_b2 = (const float*)d_in[6];
    const float* ad_W1 = (const float*)d_in[7];  const float* ad_b1 = (const float*)d_in[8];
    const float* ad_W2 = (const float*)d_in[9];  const float* ad_b2 = (const float*)d_in[10];
    const float* se_W1 = (const float*)d_in[11]; const float* se_b1 = (const float*)d_in[12];
    const float* se_W2 = (const float*)d_in[13]; const float* se_b2 = (const float*)d_in[14];
    const float* sd_W1 = (const float*)d_in[15]; const float* sd_b1 = (const float*)d_in[16];
    const float* sd_W2 = (const float*)d_in[17]; const float* sd_b2 = (const float*)d_in[18];

    // output layout: x_[N,XD], s_[N,SD], h_a[N,HID], h_s[N,HID], dna[N], dns[N]
    float* out    = (float*)d_out;
    float* x_out  = out;
    float* s_out  = out + (size_t)N_ * XD;
    float* ha_out = s_out + (size_t)N_ * SD;
    float* hs_out = ha_out + (size_t)N_ * HID;
    float* dna    = hs_out + (size_t)N_ * HID;
    float* dns    = dna + N_;

    // workspace: t1..t4 [N,HID] fp32, tot/cnt x2, then bf16 transposed weights
    float* w     = (float*)d_ws;
    float* t1    = w;
    float* t2    = t1 + (size_t)N_ * HID;
    float* t3    = t2 + (size_t)N_ * HID;
    float* t4    = t3 + (size_t)N_ * HID;
    float* tot_a = t4 + (size_t)N_ * HID;
    float* cnt_a = tot_a + N_;
    float* tot_s = cnt_a + N_;
    float* cnt_s = tot_s + N_;

    unsigned short* wt_ae1 = (unsigned short*)(cnt_s + N_);     // [HID][XD]
    unsigned short* wt_ae2 = wt_ae1 + (size_t)HID * XD;         // [HID][HID]
    unsigned short* wt_ad1 = wt_ae2 + (size_t)HID * HID;        // [HID][HID]
    unsigned short* wt_ad2 = wt_ad1 + (size_t)HID * HID;        // [XD][HID]
    unsigned short* wt_se1 = wt_ad2 + (size_t)XD * HID;         // [HID][SD]
    unsigned short* wt_se2 = wt_se1 + (size_t)HID * SD;         // [HID][HID]
    unsigned short* wt_sd1 = wt_se2 + (size_t)HID * HID;        // [HID][HID]
    unsigned short* wt_sd2 = wt_sd1 + (size_t)HID * HID;        // [SD][HID]

    dim3 blk(256);
    #define CVT_GRID(T) dim3((unsigned)(((T) + 255) / 256))
    #define GEMM_GRID(M, N) dim3((unsigned)((N) / BN), (unsigned)((M) / BM))

    // ---- pre-convert + transpose all weights to bf16 [N][K] ----
    convert_transpose_w<<<CVT_GRID(XD * HID),  blk, 0, stream>>>(ae_W1, wt_ae1, XD,  HID, XD * HID);
    convert_transpose_w<<<CVT_GRID(HID * HID), blk, 0, stream>>>(ae_W2, wt_ae2, HID, HID, HID * HID);
    convert_transpose_w<<<CVT_GRID(HID * HID), blk, 0, stream>>>(ad_W1, wt_ad1, HID, HID, HID * HID);
    convert_transpose_w<<<CVT_GRID(HID * XD),  blk, 0, stream>>>(ad_W2, wt_ad2, HID, XD,  HID * XD);
    convert_transpose_w<<<CVT_GRID(SD * HID),  blk, 0, stream>>>(se_W1, wt_se1, SD,  HID, SD * HID);
    convert_transpose_w<<<CVT_GRID(HID * HID), blk, 0, stream>>>(se_W2, wt_se2, HID, HID, HID * HID);
    convert_transpose_w<<<CVT_GRID(HID * HID), blk, 0, stream>>>(sd_W1, wt_sd1, HID, HID, HID * HID);
    convert_transpose_w<<<CVT_GRID(HID * SD),  blk, 0, stream>>>(sd_W2, wt_sd2, HID, SD,  HID * SD);

    // ---- X-side MLPs ----
    gemm_bias_act<<<GEMM_GRID(N_, HID), blk, 0, stream>>>(x,  wt_ae1, ae_b1, t1,     N_, HID, XD,  1);
    gemm_bias_act<<<GEMM_GRID(N_, HID), blk, 0, stream>>>(t1, wt_ae2, ae_b2, ha_out, N_, HID, HID, 0);
    gemm_bias_act<<<GEMM_GRID(N_, HID), blk, 0, stream>>>(ha_out, wt_ad1, ad_b1, t2, N_, HID, HID, 1);
    gemm_bias_act<<<GEMM_GRID(N_, XD),  blk, 0, stream>>>(t2, wt_ad2, ad_b2, x_out,  N_, XD,  HID, 0);

    // ---- S-side MLPs (big GEMMs) ----
    gemm_bias_act<<<GEMM_GRID(N_, HID), blk, 0, stream>>>(s,  wt_se1, se_b1, t3,     N_, HID, SD,  1);
    gemm_bias_act<<<GEMM_GRID(N_, HID), blk, 0, stream>>>(t3, wt_se2, se_b2, hs_out, N_, HID, HID, 0);
    gemm_bias_act<<<GEMM_GRID(N_, HID), blk, 0, stream>>>(hs_out, wt_sd1, sd_b1, t4, N_, HID, HID, 1);
    gemm_bias_act<<<GEMM_GRID(N_, SD),  blk, 0, stream>>>(t4, wt_sd2, sd_b2, s_out,  N_, SD,  HID, 0);

    // ---- neighbor squared-distance aggregation ----
    hipMemsetAsync(tot_a, 0, 4 * (size_t)N_ * sizeof(float), stream); // tot_a..cnt_s contiguous
    {
        const int waves_per_blk = 8;
        const unsigned nblk = (unsigned)((E + waves_per_blk - 1) / waves_per_blk);
        edge_sqdist<<<nblk, blk, 0, stream>>>(ha_out, src, dst, tot_a, cnt_a, E);
        edge_sqdist<<<nblk, blk, 0, stream>>>(hs_out, src, dst, tot_s, cnt_s, E);
    }
    finalize_mean<<<(N_ + 255) / 256, blk, 0, stream>>>(tot_a, cnt_a, dna, N_);
    finalize_mean<<<(N_ + 255) / 256, blk, 0, stream>>>(tot_s, cnt_s, dns, N_);
}
